// Conv2d_gradual_int8_27788438405485
// MI455X (gfx1250) — 8-bit matrix operand order repaired and checked by reading, compiled, not run on hardware
//
#include <hip/hip_runtime.h>
#include <hip/hip_bf16.h>
#include <stdint.h>

typedef int v8i __attribute__((ext_vector_type(8)));

#define QMAXF    127.0f
#define INV_QMAX 0.00787401574803149606f
#define B_    4
#define C_    64
#define H_    32
#define W_    32
#define O_    128
#define CK    576
#define NROWS 4096
#define KCH   9

#define WS_MAXBITS 0
#define WS_SCALEW  256
#define WS_QW      1024
#define WS_QXM     131072

#define LUT_BYTES  (256 * 256 * 4)
#define AS_OFF     LUT_BYTES
#define BS_OFF     (LUT_BYTES + 2048)
#define SMEM_BYTES (LUT_BYTES + 2048 + 4096)

__global__ void init_ws_kernel(unsigned* maxbits) {
    if (threadIdx.x == 0) *maxbits = 0u;
}

__global__ void absmax_x_kernel(const float* __restrict__ x,
                                unsigned* __restrict__ maxbits, int n) {
    __shared__ float red[256];
    float m = 0.0f;
    for (int i = blockIdx.x * blockDim.x + threadIdx.x; i < n;
         i += gridDim.x * blockDim.x)
        m = fmaxf(m, fabsf(x[i]));
    red[threadIdx.x] = m;
    __syncthreads();
    for (int s = 128; s > 0; s >>= 1) {
        if (threadIdx.x < s)
            red[threadIdx.x] = fmaxf(red[threadIdx.x], red[threadIdx.x + s]);
        __syncthreads();
    }
    if (threadIdx.x == 0) atomicMax(maxbits, __float_as_uint(red[0]));
}

__global__ void quant_w_kernel(const float* __restrict__ w,
                               float* __restrict__ scale_w,
                               signed char* __restrict__ qw) {
    const int o = blockIdx.x;
    const float* wrow = w + (size_t)o * CK;
    __shared__ float red[64];
    __shared__ float s_inv;
    float m = 0.0f;
    for (int j = threadIdx.x; j < CK; j += 64) m = fmaxf(m, fabsf(wrow[j]));
    red[threadIdx.x] = m;
    __syncthreads();
    for (int s = 32; s > 0; s >>= 1) {
        if (threadIdx.x < s)
            red[threadIdx.x] = fmaxf(red[threadIdx.x], red[threadIdx.x + s]);
        __syncthreads();
    }
    if (threadIdx.x == 0) {
        float sw = red[0] * INV_QMAX;
        scale_w[o] = sw;
        s_inv = 1.0f / sw;
    }
    __syncthreads();
    const float inv = s_inv;
    for (int j = threadIdx.x; j < CK; j += 64) {
        float q = fminf(fmaxf(rintf(wrow[j] * inv), -127.0f), 127.0f);
        qw[(size_t)o * CK + j] = (signed char)(int)q;
    }
}

__global__ void quant_im2col_kernel(const float* __restrict__ x,
                                    const unsigned* __restrict__ maxbits,
                                    signed char* __restrict__ qxm) {
    const float mx = __uint_as_float(*maxbits);
    const float sx = 0.05f * (mx * INV_QMAX) + 0.95f;
    const float inv = 1.0f / sx;
    int idx = blockIdx.x * blockDim.x + threadIdx.x;
    if (idx >= NROWS * CK) return;
    int n = idx / CK, j = idx - n * CK;
    int c = j / 9, rem = j - c * 9, kh = rem / 3, kw = rem - kh * 3;
    int b = n >> 10, oh = (n >> 5) & 31, ow = n & 31;
    int ih = oh + kh - 1, iw = ow + kw - 1;
    signed char q = 0;
    if ((unsigned)ih < (unsigned)H_ && (unsigned)iw < (unsigned)W_) {
        float v = x[(((size_t)b * C_ + c) * H_ + ih) * W_ + iw];
        q = (signed char)(int)fminf(fmaxf(rintf(v * inv), -127.0f), 127.0f);
    }
    qxm[idx] = q;
}

__global__ __launch_bounds__(256, 1)
void gemm_lut_kernel(const signed char* __restrict__ qxm,
                     const signed char* __restrict__ qw,
                     const float* __restrict__ lut,
                     const float* __restrict__ scale_w,
                     const unsigned* __restrict__ maxbits,
                     const float* __restrict__ bias,
                     float* __restrict__ out) {
    extern __shared__ char smem[];
    float*       lut_s = (float*)smem;
    signed char* As    = (signed char*)(smem + AS_OFF);
    signed char* Bs    = (signed char*)(smem + BS_OFF);

    const int tid    = threadIdx.x;
    const int lane   = tid & 31;
    const int wave   = tid >> 5;
    const int wave_m = wave >> 2;
    const int wave_o = wave & 3;
    const int n_base = blockIdx.x * 32;
    const int o_base = blockIdx.y * 64;
    const int half   = lane >> 4;
    const int l15    = lane & 15;

    {
        const float4* src = (const float4*)lut;
        float4*       dst = (float4*)lut_s;
        for (int i = tid; i < (256 * 256) / 4; i += 256) dst[i] = src[i];
    }

    const float mx = __uint_as_float(*maxbits);
    const float sx = 0.05f * (mx * INV_QMAX) + 0.95f;

    v8i   acc = {};
    float facc[8];
#pragma unroll
    for (int r = 0; r < 8; ++r) facc[r] = 0.0f;

    const int arow = wave_m * 16 + l15;
    const int orow = wave_o * 16 + l15;

    __syncthreads();

    for (int kc = 0; kc < KCH; ++kc) {
        const int k0 = kc * 64;
        for (int i = tid; i < 512; i += 256) {
            int r = i >> 4, c4 = i & 15;
            ((int*)As)[i] =
                *(const int*)(qxm + (size_t)(n_base + r) * CK + k0 + c4 * 4);
        }
        for (int i = tid; i < 1024; i += 256) {
            int r = i >> 4, c4 = i & 15;
            ((int*)Bs)[i] =
                *(const int*)(qw + (size_t)(o_base + r) * CK + k0 + c4 * 4);
        }
        if (kc + 1 < KCH)
            __builtin_prefetch(qxm + (size_t)n_base * CK + (k0 + 64), 0, 1);
        __syncthreads();

        v8i a, bf;
        {
            const signed char* ap = As + arow * 64 + half * 16;
            a[0] = *(const int*)(ap + 0);  a[1] = *(const int*)(ap + 4);
            a[2] = *(const int*)(ap + 8);  a[3] = *(const int*)(ap + 12);
            a[4] = *(const int*)(ap + 32); a[5] = *(const int*)(ap + 36);
            a[6] = *(const int*)(ap + 40); a[7] = *(const int*)(ap + 44);
            const signed char* bp = Bs + orow * 64 + half * 16;
            bf[0] = *(const int*)(bp + 0);  bf[1] = *(const int*)(bp + 4);
            bf[2] = *(const int*)(bp + 8);  bf[3] = *(const int*)(bp + 12);
            bf[4] = *(const int*)(bp + 32); bf[5] = *(const int*)(bp + 36);
            bf[6] = *(const int*)(bp + 40); bf[7] = *(const int*)(bp + 44);
        }
        acc = __builtin_amdgcn_wmma_i32_16x16x64_iu8(
             true, a,  true, bf, acc,
             false,  false);

        const signed char* brow  = Bs + orow * 64;
        const signed char* arows = As + (wave_m * 16 + half * 8) * 64;
        for (int jg = 0; jg < 64; jg += 4) {
            const unsigned bq4 = *(const unsigned*)(brow + jg);
            unsigned adw[8];
#pragma unroll
            for (int r = 0; r < 8; ++r)
                adw[r] = *(const unsigned*)(arows + r * 64 + jg);
#pragma unroll
            for (int t = 0; t < 4; ++t) {
                const unsigned sel =
                    0x0C0C0000u | ((unsigned)(4 + t) << 8) | (unsigned)t;
#pragma unroll
                for (int r = 0; r < 8; ++r) {
                    unsigned idx =
                        __builtin_amdgcn_perm(adw[r], bq4, sel) ^ 0x8080u;
                    facc[r] += lut_s[idx];
                }
            }
        }
        __syncthreads();
    }

    const int   o_g   = o_base + orow;
    const float sw    = scale_w[o_g];
    const float bo    = bias[o_g];
    const float scale = sx * sw;
#pragma unroll
    for (int r = 0; r < 8; ++r) {
        int   n  = n_base + wave_m * 16 + half * 8 + r;
        int   b  = n >> 10, oh = (n >> 5) & 31, ow = n & 31;
        float v  = (0.5f * (float)acc[r] + 0.5f * facc[r]) * scale + bo;
        out[(((size_t)b * O_ + o_g) * H_ + oh) * W_ + ow] = v;
    }
}

extern "C" void kernel_launch(void* const* d_in, const int* in_sizes, int n_in,
                              void* d_out, int out_size, void* d_ws, size_t ws_size,
                              hipStream_t stream) {
    (void)in_sizes; (void)n_in; (void)out_size; (void)ws_size;
    const float* x    = (const float*)d_in[0];
    const float* w    = (const float*)d_in[1];
    const float* bias = (const float*)d_in[2];
    const float* lut  = (const float*)d_in[3];
    float* out = (float*)d_out;

    char* ws = (char*)d_ws;
    unsigned*    maxbits = (unsigned*)(ws + WS_MAXBITS);
    float*       scale_w = (float*)(ws + WS_SCALEW);
    signed char* qw      = (signed char*)(ws + WS_QW);
    signed char* qxm     = (signed char*)(ws + WS_QXM);

    init_ws_kernel<<<1, 32, 0, stream>>>(maxbits);
    absmax_x_kernel<<<256, 256, 0, stream>>>(x, maxbits, B_ * C_ * H_ * W_);
    quant_w_kernel<<<O_, 64, 0, stream>>>(w, scale_w, qw);
    quant_im2col_kernel<<<(NROWS * CK + 255) / 256, 256, 0, stream>>>(x, maxbits, qxm);

    (void)hipFuncSetAttribute((const void*)gemm_lut_kernel,
                              hipFuncAttributeMaxDynamicSharedMemorySize,
                              SMEM_BYTES);
    dim3 grid(NROWS / 32, O_ / 64);
    gemm_lut_kernel<<<grid, 256, SMEM_BYTES, stream>>>(
        qxm, qw, lut, scale_w, maxbits, bias, out);
}
